// BidirectionalTemporalAttention_16604343566617
// MI455X (gfx1250) — matrix operand order repaired and checked by reading, not compiled, not run on hardware
//
#include <hip/hip_runtime.h>
#include <hip/hip_bf16.h>

// ---------------------------------------------------------------------------
// BidirectionalTemporalAttention on gfx1250 (MI455X), bf16 WMMA pipeline.
//   B=2, N=2048, D=1024, H=16 (heads 0-7 lookbehind/causal, 8-15 lookahead),
//   E=64. All matmuls on v_wmma_f32_16x16x32_bf16, f32 accumulate.
// Stages: cast->bf16 | QKV WMMA GEMMs (V stored transposed) | fused
// flash-style masked attention | out-projection WMMA GEMM -> fp32 d_out.
// Workspace ~48 MB (layout below). Wave-uniform scalars are forced into
// SGPRs via readfirstlane so all WMMA executes under scalar control flow
// (EXEC all-ones, per ISA 7.12 requirement).
// ---------------------------------------------------------------------------

typedef __attribute__((ext_vector_type(16))) __bf16 v16bf;
typedef __attribute__((ext_vector_type(8)))  float  v8f;

union AB16 { v16bf v; uint4 q[2]; unsigned short h[16]; };

#define NEG_INF (-3.0e38f)

__device__ __forceinline__ unsigned short f2bfu(float f) {
  unsigned u = __builtin_bit_cast(unsigned, f);
  unsigned r = u + 0x7FFFu + ((u >> 16) & 1u);
  return (unsigned short)(r >> 16);
}

// xor-lane reduction step via ds_swizzle (group-of-32: and=0x1f, xor=MASK).
#if __has_builtin(__builtin_amdgcn_ds_swizzle)
template <int MASK>
__device__ __forceinline__ float swz_xor(float x) {
  int i = __builtin_amdgcn_ds_swizzle(__builtin_bit_cast(int, x),
                                      (MASK << 10) | 0x1F);
  return __builtin_bit_cast(float, i);
}
#else
template <int MASK>
__device__ __forceinline__ float swz_xor(float x) {
  return __shfl_xor(x, MASK, 32);
}
#endif

__device__ __forceinline__ float red_max16(float x) {
  x = fmaxf(x, swz_xor<1>(x));
  x = fmaxf(x, swz_xor<2>(x));
  x = fmaxf(x, swz_xor<4>(x));
  x = fmaxf(x, swz_xor<8>(x));
  return x;
}
__device__ __forceinline__ float red_sum16(float x) {
  x += swz_xor<1>(x);
  x += swz_xor<2>(x);
  x += swz_xor<4>(x);
  x += swz_xor<8>(x);
  return x;
}

// A-matrix 16x32 bf16 fragment from row-major src (lda in elements).
// Per ISA 7.12.2: lane L -> row M = L%16; halves 0..7 hold K = 8*(L/16)+i,
// halves 8..15 hold K = 16 + 8*(L/16) + (i-8): two contiguous b128 loads.
__device__ __forceinline__ v16bf load_frag_a(const unsigned short* p, int lda,
                                             int m0, int k0, int lane) {
  const unsigned short* q =
      p + (size_t)(m0 + (lane & 15)) * lda + k0 + ((lane >> 4) << 3);
  AB16 f;
  f.q[0] = *(const uint4*)(q);
  f.q[1] = *(const uint4*)(q + 16);
  return f.v;
}

// B-matrix 32x16 bf16 fragment where the logical B = Bw^T for row-major
// Bw (N x K): lane L -> column N = L%16; halves 0..7 hold K = 8*(L/16)+i,
// halves 8..15 hold K = 16 + 8*(L/16) + (i-8): two b128 loads.
__device__ __forceinline__ v16bf load_frag_bt(const unsigned short* p, int ldb,
                                              int n0, int k0, int lane) {
  const unsigned short* q =
      p + (size_t)(n0 + (lane & 15)) * ldb + k0 + ((lane >> 4) << 3);
  AB16 f;
  f.q[0] = *(const uint4*)(q);
  f.q[1] = *(const uint4*)(q + 16);
  return f.v;
}

__device__ __forceinline__ v8f wmma_bf16(v16bf a, v16bf b, v8f c) {
  return __builtin_amdgcn_wmma_f32_16x16x32_bf16(
      /*neg_a=*/false, a, /*neg_b=*/false, b,
      /*c_mod=*/(short)0, c, /*reuse_a=*/false, /*reuse_b=*/false);
}

// ---------------------------------------------------------------------------
// fp32 -> bf16 cast, 4 elements/thread.
__global__ __launch_bounds__(256) void cast_f32_bf16(
    const float* __restrict__ src, unsigned short* __restrict__ dst, int n) {
  int i = (blockIdx.x * blockDim.x + threadIdx.x) * 4;
  if (i >= n) return;
  float4 f = *(const float4*)(src + i);
  uint2 o;
  o.x = (unsigned)f2bfu(f.x) | ((unsigned)f2bfu(f.y) << 16);
  o.y = (unsigned)f2bfu(f.z) | ((unsigned)f2bfu(f.w) << 16);
  *(uint2*)(dst + i) = o;
}

// ---------------------------------------------------------------------------
// Store one 16x16 f32 accumulator tile.
// MODE 0: bf16 row-major; MODE 1: bf16 "Vt" transposed per b-slice of 2048
// rows: addr = (m/2048)*2097152 + col*2048 + (m%2048); MODE 2: f32 row-major.
template <int MODE>
__device__ __forceinline__ void store_tile(void* Cout, v8f acc, int m0, int n0,
                                           int ldc, int lane) {
  if (MODE == 0) {
    unsigned short* C = (unsigned short*)Cout;
#pragma unroll
    for (int r = 0; r < 8; ++r) {
      int m = m0 + r + ((lane >> 4) << 3);
      C[(size_t)m * ldc + n0 + (lane & 15)] = f2bfu(acc[r]);
    }
  } else if (MODE == 1) {
    unsigned short* C = (unsigned short*)Cout;
    int bidx = m0 >> 11;
    int col = n0 + (lane & 15);
    int nrow = (m0 & 2047) + ((lane >> 4) << 3);
    union { unsigned short h[8]; uint4 q; } t;
#pragma unroll
    for (int r = 0; r < 8; ++r) t.h[r] = f2bfu(acc[r]);
    *(uint4*)(C + (size_t)bidx * 2097152 + (size_t)col * 2048 + nrow) = t.q;
  } else {
    float* C = (float*)Cout;
#pragma unroll
    for (int r = 0; r < 8; ++r) {
      int m = m0 + r + ((lane >> 4) << 3);
      C[(size_t)m * ldc + n0 + (lane & 15)] = acc[r];
    }
  }
}

// ---------------------------------------------------------------------------
// WMMA GEMM: C[M x N] = A[M x K] (row-major bf16) @ Bw[N x K]^T.
// One wave computes a 32x32 tile (2x2 WMMA tiles): 4 wmma per 8 b128 loads,
// each fragment reused twice. K-loop in steps of 32 under scalar control.
template <int MODE>
__global__ __launch_bounds__(256) void gemm_wmma(
    const unsigned short* __restrict__ A, const unsigned short* __restrict__ Bw,
    void* __restrict__ Cout, int M, int N, int K, int lda, int ldb, int ldc) {
  int wave = __builtin_amdgcn_readfirstlane(
      (int)((blockIdx.x * blockDim.x + threadIdx.x) >> 5));
  int lane = threadIdx.x & 31;
  int tilesN = N >> 5;
  int tm = wave / tilesN;
  int tn = wave - tm * tilesN;
  if (tm >= (M >> 5)) return;  // scalar branch
  int m0 = tm << 5, n0 = tn << 5;

  const unsigned short* arow = A + (size_t)(m0 + (lane & 15)) * lda;
  const unsigned short* brow = Bw + (size_t)(n0 + (lane & 15)) * ldb;

  v8f acc00 = {0.f,0.f,0.f,0.f,0.f,0.f,0.f,0.f};
  v8f acc01 = acc00, acc10 = acc00, acc11 = acc00;
  for (int k0 = 0; k0 < K; k0 += 32) {
    if (k0 + 32 < K) {  // global_prefetch_b8 of next k-slice
      __builtin_prefetch(arow + k0 + 32, 0, 1);
      __builtin_prefetch(brow + k0 + 32, 0, 1);
    }
    v16bf a0 = load_frag_a(A, lda, m0, k0, lane);
    v16bf a1 = load_frag_a(A, lda, m0 + 16, k0, lane);
    v16bf b0 = load_frag_bt(Bw, ldb, n0, k0, lane);
    v16bf b1 = load_frag_bt(Bw, ldb, n0 + 16, k0, lane);
    acc00 = wmma_bf16(a0, b0, acc00);
    acc01 = wmma_bf16(a0, b1, acc01);
    acc10 = wmma_bf16(a1, b0, acc10);
    acc11 = wmma_bf16(a1, b1, acc11);
  }
  store_tile<MODE>(Cout, acc00, m0,      n0,      ldc, lane);
  store_tile<MODE>(Cout, acc01, m0,      n0 + 16, ldc, lane);
  store_tile<MODE>(Cout, acc10, m0 + 16, n0,      ldc, lane);
  store_tile<MODE>(Cout, acc11, m0 + 16, n0 + 16, ldc, lane);
}

// ---------------------------------------------------------------------------
// Fused flash-style masked attention. One wave handles one (b, h, 16-query
// tile); streams 32-key chunks with online softmax. Heads 0-7: causal
// (keep col<=row); heads 8-15: anti-causal (keep col>=row). Fully masked
// chunks are skipped via the scalar chunk range.
__global__ __launch_bounds__(256) void attn_wmma(
    const unsigned short* __restrict__ Qb, const unsigned short* __restrict__ Kb,
    const unsigned short* __restrict__ Vt, unsigned short* __restrict__ Obuf) {
  __shared__ unsigned short plds[8][16][32];  // per-wave P tile bounce (8 KB)

  int wave = __builtin_amdgcn_readfirstlane(
      (int)((blockIdx.x * blockDim.x + threadIdx.x) >> 5));
  int w = __builtin_amdgcn_readfirstlane((int)(threadIdx.x >> 5));
  int lane = threadIdx.x & 31;
  int b = wave >> 11;          // / (16*128)   (scalar)
  int h = (wave >> 7) & 15;    // (scalar)
  int qt = wave & 127;         // query tile of 16 (scalar)
  bool lb = (h < 8);           // scalar branch condition

  const unsigned short* Qp = Qb + ((size_t)b << 21) + h * 64;
  const unsigned short* Kp = Kb + ((size_t)b << 21) + h * 64;
  const unsigned short* Vp = Vt + ((size_t)b << 21) + ((size_t)(h * 64) << 11);

  // Q A-fragments for e = 0..31 and 32..63 (K dim of the QK^T GEMM)
  v16bf qa0 = load_frag_a(Qp, 1024, qt * 16, 0, lane);
  v16bf qa1 = load_frag_a(Qp, 1024, qt * 16, 32, lane);

  v8f o0 = {0.f,0.f,0.f,0.f,0.f,0.f,0.f,0.f};
  v8f o1 = o0, o2v = o0, o3 = o0;
  float mrow[8], lrow[8];
#pragma unroll
  for (int r = 0; r < 8; ++r) { mrow[r] = NEG_INF; lrow[r] = 0.f; }

  int c0 = lb ? 0 : (qt >> 1);          // scalar
  int c1 = lb ? (qt >> 1) : 63;         // scalar
  int rowq0 = qt * 16 + ((lane >> 4) << 3);
  int colL = lane & 15;

  for (int c = c0; c <= c1; ++c) {      // scalar loop
    int kc = c << 5;
    // S = Q K^T for keys [kc, kc+32): two 16x16 tiles, K(=E) = 64 -> 2 wmma ea
    v8f s0 = {0.f,0.f,0.f,0.f,0.f,0.f,0.f,0.f};
    v8f s1 = s0;
    s0 = wmma_bf16(qa0, load_frag_bt(Kp, 1024, kc, 0, lane), s0);
    s0 = wmma_bf16(qa1, load_frag_bt(Kp, 1024, kc, 32, lane), s0);
    s1 = wmma_bf16(qa0, load_frag_bt(Kp, 1024, kc + 16, 0, lane), s1);
    s1 = wmma_bf16(qa1, load_frag_bt(Kp, 1024, kc + 16, 32, lane), s1);

    int col0 = kc + colL;
#pragma unroll
    for (int r = 0; r < 8; ++r) {
      float v0 = s0[r] * 0.125f;  // 1/sqrt(64)
      float v1 = s1[r] * 0.125f;
      int row = rowq0 + r;
      if (lb) {
        if (col0 > row) v0 = NEG_INF;
        if (col0 + 16 > row) v1 = NEG_INF;
      } else {
        if (col0 < row) v0 = NEG_INF;
        if (col0 + 16 < row) v1 = NEG_INF;
      }
      // row max/sum over the 16 lanes holding this row (xor<16 stays in-half)
      float mx = red_max16(fmaxf(v0, v1));
      float nm = fmaxf(mrow[r], mx);
      float sc = __expf(mrow[r] - nm);
      mrow[r] = nm;
      float p0 = __expf(v0 - nm);
      float p1 = __expf(v1 - nm);
      float ps = red_sum16(p0 + p1);
      lrow[r] = lrow[r] * sc + ps;
      o0[r] *= sc; o1[r] *= sc; o2v[r] *= sc; o3[r] *= sc;
      plds[w][row - qt * 16][colL] = f2bfu(p0);
      plds[w][row - qt * 16][colL + 16] = f2bfu(p1);
    }
    // C-frag -> A-frag relayout through wave-private LDS
    asm volatile("s_wait_dscnt 0" ::: "memory");
    v16bf pa = load_frag_a(&plds[w][0][0], 32, 0, 0, lane);
    // O += P @ V : B-frag element [k][e] = Vt[h*64+e][kc+k] (contiguous)
    o0  = wmma_bf16(pa, load_frag_bt(Vp, 2048, 0,  kc, lane), o0);
    o1  = wmma_bf16(pa, load_frag_bt(Vp, 2048, 16, kc, lane), o1);
    o2v = wmma_bf16(pa, load_frag_bt(Vp, 2048, 32, kc, lane), o2v);
    o3  = wmma_bf16(pa, load_frag_bt(Vp, 2048, 48, kc, lane), o3);
  }

  // Epilogue: normalize and store [b][h][n][e] bf16 (== reference's raw
  // (B,H,N,E)->(B,N,D) reshape on the flat buffer).
  unsigned short* Op = Obuf + ((size_t)(b * 16 + h) << 17);
#pragma unroll
  for (int r = 0; r < 8; ++r) {
    float inv = 1.0f / lrow[r];
    int n = qt * 16 + r + ((lane >> 4) << 3);
    unsigned short* dst = Op + (size_t)n * 64 + colL;
    dst[0]  = f2bfu(o0[r] * inv);
    dst[16] = f2bfu(o1[r] * inv);
    dst[32] = f2bfu(o2v[r] * inv);
    dst[48] = f2bfu(o3[r] * inv);
  }
}

// ---------------------------------------------------------------------------
extern "C" void kernel_launch(void* const* d_in, const int* in_sizes, int n_in,
                              void* d_out, int out_size, void* d_ws,
                              size_t ws_size, hipStream_t stream) {
  (void)in_sizes; (void)n_in; (void)out_size; (void)ws_size;
  const float* x     = (const float*)d_in[0];
  const float* Wq_lb = (const float*)d_in[1];
  const float* Wk_lb = (const float*)d_in[2];
  const float* Wv_lb = (const float*)d_in[3];
  const float* Wq_la = (const float*)d_in[4];
  const float* Wk_la = (const float*)d_in[5];
  const float* Wv_la = (const float*)d_in[6];
  const float* Wo    = (const float*)d_in[7];

  unsigned short* ws   = (unsigned short*)d_ws;
  unsigned short* xb   = ws;                  // 4194304 elems
  unsigned short* WqA  = xb + 4194304;        // 1048576
  unsigned short* WkA  = WqA + 1048576;
  unsigned short* WvA  = WkA + 1048576;
  unsigned short* WoB  = WvA + 1048576;
  unsigned short* Qb   = WoB + 1048576;       // 4194304
  unsigned short* Kb   = Qb + 4194304;
  unsigned short* Vt   = Kb + 4194304;
  unsigned short* Obuf = Vt + 4194304;

  // Stage 1: cast everything to bf16 (head order lb || la matches concat).
  cast_f32_bf16<<<4096, 256, 0, stream>>>(x, xb, 4194304);
  cast_f32_bf16<<<512, 256, 0, stream>>>(Wq_lb, WqA, 524288);
  cast_f32_bf16<<<512, 256, 0, stream>>>(Wq_la, WqA + 524288, 524288);
  cast_f32_bf16<<<512, 256, 0, stream>>>(Wk_lb, WkA, 524288);
  cast_f32_bf16<<<512, 256, 0, stream>>>(Wk_la, WkA + 524288, 524288);
  cast_f32_bf16<<<512, 256, 0, stream>>>(Wv_lb, WvA, 524288);
  cast_f32_bf16<<<512, 256, 0, stream>>>(Wv_la, WvA + 524288, 524288);
  cast_f32_bf16<<<1024, 256, 0, stream>>>(Wo, WoB, 1048576);

  // Stage 2: QKV projections (4096x1024x1024 each): 32x32 tile per wave,
  // (4096/32)*(1024/32) = 4096 waves -> 512 blocks.
  gemm_wmma<0><<<512, 256, 0, stream>>>(xb, WqA, Qb, 4096, 1024, 1024, 1024, 1024, 1024);
  gemm_wmma<0><<<512, 256, 0, stream>>>(xb, WkA, Kb, 4096, 1024, 1024, 1024, 1024, 1024);
  gemm_wmma<1><<<512, 256, 0, stream>>>(xb, WvA, Vt, 4096, 1024, 1024, 1024, 1024, 0);

  // Stage 3: fused masked attention. 2*16*128 = 4096 waves -> 512 blocks.
  attn_wmma<<<512, 256, 0, stream>>>(Qb, Kb, Vt, Obuf);

  // Stage 4: output projection, fp32 result straight to d_out.
  gemm_wmma<2><<<512, 256, 0, stream>>>(Obuf, WoB, (float*)d_out, 4096, 1024,
                                        1024, 1024, 1024, 1024);
}
